// EfficientMultiHeadAttention_21680994910470
// MI455X (gfx1250) — matrix operand order repaired and checked by reading, not compiled, not run on hardware
//
#include <hip/hip_runtime.h>
#include <hip/hip_bf16.h>

// ---------------- problem constants ----------------
#define B_   2
#define NN_  32768            // H*W*D tokens per batch
#define C_   256
#define NH_  8
#define DH_  32
#define M_   (B_ * NN_)       // 65536 rows
#define FC_  (4 * C_)         // 1024
#define LN_EPS 1e-3f

typedef __attribute__((ext_vector_type(16))) _Float16 v16h;
typedef __attribute__((ext_vector_type(8)))  _Float16 v8h;
typedef __attribute__((ext_vector_type(8)))  float    v8f;

// ---------------- WMMA helpers (CDNA5 layouts, ISA 7.12.2) ----------------
__device__ __forceinline__ v8f wmma_f16(v16h a, v16h b, v8f c) {
  return __builtin_amdgcn_wmma_f32_16x16x32_f16(false, a, false, b, (short)0, c,
                                                false, false);
}

// A 16x32 f16 tile (ld = 64 halves in LDS): lane=(h<<4)|r holds row m=r,
// element e -> k = 8h + (e&7) + (e&8?16:0)
__device__ __forceinline__ v16h load_fragA(const _Float16* row_k0, int h) {
  v8h lo = *(const v8h*)(row_k0 + 8 * h);        // k = 8h .. 8h+7
  v8h hi = *(const v8h*)(row_k0 + 8 * h + 16);   // k = 8h+16 .. 8h+23
  v16h f;
#pragma unroll
  for (int i = 0; i < 8; ++i) { f[i] = lo[i]; f[i + 8] = hi[i]; }
  return f;
}

// B 32x16 f16 tile from WT rows (ld = 64): lane n=r, element e -> k = 8h + (e&7) + (e&8?16:0)
__device__ __forceinline__ v16h load_fragB(const _Float16* row_k0, int h) {
  v8h lo = *(const v8h*)(row_k0 + 8 * h);
  v8h hi = *(const v8h*)(row_k0 + 8 * h + 16);
  v16h f;
#pragma unroll
  for (int i = 0; i < 8; ++i) { f[i] = lo[i]; f[i + 8] = hi[i]; }
  return f;
}

// async 16-byte copy global -> LDS (per lane), tracked by ASYNCcnt
__device__ __forceinline__ void async_cp16(unsigned lds_addr, unsigned goff,
                                           const _Float16* base) {
  asm volatile("global_load_async_to_lds_b128 %0, %1, %2"
               :: "v"(lds_addr), "v"(goff), "s"(base)
               : "memory");
}

// ---------------- GEMM: out[M x N] = A[M x K](f16) * WT[N x K](f16)^T + bias -----
// Async-LDS double-buffered pipeline. Block tile 128x64, K-stage 64.
// MODE 0: fp32 out + bias.  MODE 1: bias + exact GELU -> f16 out.
template <int MODE>
__global__ __launch_bounds__(256) void gemm_wmma(
    const _Float16* __restrict__ A, const _Float16* __restrict__ WT,
    const float* __restrict__ bias, float* __restrict__ outF,
    _Float16* __restrict__ outH, int Ndim, int Kdim) {
  __shared__ alignas(16) _Float16 As[2][128 * 64];   // 2 x 16 KB
  __shared__ alignas(16) _Float16 Bs[2][64 * 64];    // 2 x 8 KB

  const int tid  = threadIdx.x;
  const int lane = tid & 31;
  const int wid  = tid >> 5;                  // 0..7
  const int wr   = wid >> 1, wc = wid & 1;    // 4x2 wave grid -> 128x64 block tile
  const int row0 = blockIdx.x * 128;
  const int col0 = blockIdx.y * 64;
  const int r = lane & 15, h = lane >> 4;

  // ---- async staging setup: A stage = 1024 16B chunks, B stage = 512 chunks ----
  const unsigned aBase = (unsigned)(size_t)&As[0][0];
  const unsigned bBase = (unsigned)(size_t)&Bs[0][0];
  unsigned ga[4], la[4], gb[2], lb[2];
#pragma unroll
  for (int j = 0; j < 4; ++j) {
    int c = tid + 256 * j;                    // 0..1023
    int rowc = c >> 3, c16 = c & 7;           // row 0..127, 16B column
    ga[j] = (unsigned)(((size_t)(row0 + rowc) * Kdim + c16 * 8) * 2);
    la[j] = aBase + (unsigned)((rowc * 64 + c16 * 8) * 2);
  }
#pragma unroll
  for (int j = 0; j < 2; ++j) {
    int c = tid + 256 * j;                    // 0..511
    int nc = c >> 3, c16 = c & 7;             // n 0..63
    gb[j] = (unsigned)(((size_t)(col0 + nc) * Kdim + c16 * 8) * 2);
    lb[j] = bBase + (unsigned)((nc * 64 + c16 * 8) * 2);
  }

  auto issue_stage = [&](int bsel) {
    const unsigned at = bsel ? (unsigned)(128 * 64 * 2) : 0u;
    const unsigned bt = bsel ? (unsigned)(64 * 64 * 2) : 0u;
#pragma unroll
    for (int j = 0; j < 4; ++j) async_cp16(la[j] + at, ga[j], A);
#pragma unroll
    for (int j = 0; j < 2; ++j) async_cp16(lb[j] + bt, gb[j], WT);
#pragma unroll
    for (int j = 0; j < 4; ++j) ga[j] += 64 * 2;   // advance one K-stage
#pragma unroll
    for (int j = 0; j < 2; ++j) gb[j] += 64 * 2;
  };

  v8f acc00 = {}, acc01 = {}, acc10 = {}, acc11 = {};

  issue_stage(0);                              // prologue: stage 0 -> buf 0
  const int nstages = Kdim >> 6;               // K / 64
  int buf = 0;
  for (int s = 0; s < nstages; ++s) {
    if (s + 1 < nstages) {
      issue_stage(buf ^ 1);                    // prefetch next stage
      asm volatile("s_wait_asynccnt 0x6" ::: "memory");  // prev 6 retired (in-order)
    } else {
      asm volatile("s_wait_asynccnt 0x0" ::: "memory");
    }
    __syncthreads();

    const _Float16* Ab = &As[buf][0];
    const _Float16* Bb = &Bs[buf][0];
#pragma unroll
    for (int kk = 0; kk < 64; kk += 32) {
      v16h a0 = load_fragA(Ab + (wr * 32 + r)      * 64 + kk, h);
      v16h a1 = load_fragA(Ab + (wr * 32 + 16 + r) * 64 + kk, h);
      v16h b0 = load_fragB(Bb + (wc * 32 + r)      * 64 + kk, h);
      v16h b1 = load_fragB(Bb + (wc * 32 + 16 + r) * 64 + kk, h);
      acc00 = wmma_f16(a0, b0, acc00);
      acc01 = wmma_f16(a0, b1, acc01);
      acc10 = wmma_f16(a1, b0, acc10);
      acc11 = wmma_f16(a1, b1, acc11);
    }
    __syncthreads();                           // buffer safe to overwrite
    buf ^= 1;
  }

  // ---- epilogue: C/D layout lane n = r, VGPR v -> m = v + 8h ----
  v8f accs[2][2] = {{acc00, acc01}, {acc10, acc11}};
#pragma unroll
  for (int i = 0; i < 2; ++i)
#pragma unroll
    for (int j = 0; j < 2; ++j) {
      int n     = col0 + wc * 32 + j * 16 + r;
      int mbase = row0 + wr * 32 + i * 16 + h * 8;
      float bv  = bias[n];
#pragma unroll
      for (int v = 0; v < 8; ++v) {
        float val  = accs[i][j][v] + bv;
        size_t idx = (size_t)(mbase + v) * Ndim + n;
        if (MODE == 1) {
          float g = 0.5f * val * (1.0f + erff(val * 0.70710678118654752f));
          outH[idx] = (_Float16)g;
        } else {
          outF[idx] = val;
        }
      }
    }
}

// ---------------- conversion kernels ----------------
__global__ void cvt_f32_f16(const float* __restrict__ in,
                            _Float16* __restrict__ out, size_t n) {
  size_t i = (size_t)blockIdx.x * blockDim.x + threadIdx.x;
  size_t stride = (size_t)gridDim.x * blockDim.x;
  for (; i < n; i += stride) out[i] = (_Float16)in[i];
}

// in: K x N row-major fp32; out: N x K row-major f16 (transposed)
__global__ void transpose_cvt(const float* __restrict__ in,
                              _Float16* __restrict__ out, int K, int N) {
  int idx = blockIdx.x * 256 + threadIdx.x;
  if (idx >= K * N) return;
  int n = idx / K, k = idx - n * K;
  out[idx] = (_Float16)in[(size_t)k * N + n];
}

// ---------------- attention statistics ----------------
__global__ void init_stats(float* colmax, float* colsum, float* ctxraw) {
  int i = blockIdx.x * 256 + threadIdx.x;
  if (i < B_ * C_) { colmax[i] = -INFINITY; colsum[i] = 0.f; }
  if (i < B_ * NH_ * DH_ * DH_) ctxraw[i] = 0.f;
}

__device__ __forceinline__ void atomicMaxFloat(float* addr, float val) {
  if (val >= 0.f)
    atomicMax((int*)addr, __float_as_int(val));
  else
    atomicMin((unsigned int*)addr, (unsigned int)__float_as_int(val));
}

// column max of Kp over token axis; block = (b, 128-row chunk), thread = channel
__global__ __launch_bounds__(256) void colmax_k(const float* __restrict__ Kp,
                                                float* __restrict__ colmax) {
  const int chunks = NN_ / 128;
  int b = blockIdx.x / chunks, ch = blockIdx.x % chunks;
  int c = threadIdx.x;
  size_t base = ((size_t)b * NN_ + (size_t)ch * 128) * C_ + c;
  float m = -INFINITY;
  for (int i = 0; i < 128; ++i) m = fmaxf(m, Kp[base + (size_t)i * C_]);
  atomicMaxFloat(&colmax[b * C_ + c], m);
}

// Per (b,h,512-token chunk): accumulate colsum[d] = sum_n exp(k-max) and
// ctxraw[d][e] = sum_n exp(k-max)[d] * v[e].
__global__ __launch_bounds__(256) void ctx_accum(
    const float* __restrict__ Kp, const float* __restrict__ Vp,
    const float* __restrict__ colmax, float* __restrict__ colsum,
    float* __restrict__ ctxraw) {
  const int CHUNK = 512;
  const int chunks = NN_ / CHUNK;
  int chunk = blockIdx.x % chunks;
  int bh    = blockIdx.x / chunks;
  int b = bh / NH_, hH = bh % NH_;

  __shared__ float ek[8][DH_];
  __shared__ float vv[8][DH_];

  int tid = threadIdx.x;
  int d   = tid & 31;       // load channel / colsum channel
  int nl  = tid >> 5;       // 0..7 local row
  float cmax = colmax[b * C_ + hH * DH_ + d];

  float a0 = 0.f, a1 = 0.f, a2 = 0.f, a3 = 0.f, ssum = 0.f;
  int e  = tid & 31;        // accumulation column
  int d0 = tid >> 5;        // accumulation row base (d0, d0+8, d0+16, d0+24)

  int nbeg = chunk * CHUNK;
  for (int n0 = nbeg; n0 < nbeg + CHUNK; n0 += 8) {
    size_t base = ((size_t)b * NN_ + (n0 + nl)) * C_ + hH * DH_;
    float ex = __expf(Kp[base + d] - cmax);
    ek[nl][d] = ex;
    ssum += ex;
    vv[nl][d] = Vp[base + d];
    __syncthreads();
#pragma unroll
    for (int nn = 0; nn < 8; ++nn) {
      float v0 = vv[nn][e];
      a0 = fmaf(ek[nn][d0],      v0, a0);
      a1 = fmaf(ek[nn][d0 + 8],  v0, a1);
      a2 = fmaf(ek[nn][d0 + 16], v0, a2);
      a3 = fmaf(ek[nn][d0 + 24], v0, a3);
    }
    __syncthreads();
  }
  atomicAdd(&colsum[b * C_ + hH * DH_ + d], ssum);
  float* cb = ctxraw + ((size_t)(b * NH_ + hH) * DH_) * DH_;
  atomicAdd(&cb[(d0)      * DH_ + e], a0);
  atomicAdd(&cb[(d0 + 8)  * DH_ + e], a1);
  atomicAdd(&cb[(d0 + 16) * DH_ + e], a2);
  atomicAdd(&cb[(d0 + 24) * DH_ + e], a3);
}

__global__ void ctx_final(const float* __restrict__ ctxraw,
                          const float* __restrict__ colsum,
                          float* __restrict__ ctx) {
  int i = blockIdx.x * 256 + threadIdx.x;
  if (i >= B_ * NH_ * DH_ * DH_) return;
  int d  = (i >> 5) & 31;
  int bh = i >> 10;
  int b  = bh >> 3, hH = bh & 7;
  ctx[i] = ctxraw[i] / colsum[b * C_ + hH * DH_ + d];
}

// ---------------- per-token: q-softmax, apply context, residual + LN1 ----------------
__global__ __launch_bounds__(256) void attn_ln1(
    const float* __restrict__ inq, const float* __restrict__ Pq,
    const float* __restrict__ ctx, const float* __restrict__ g,
    const float* __restrict__ bta, float* __restrict__ X,
    _Float16* __restrict__ X16) {
  int token = blockIdx.x;               // 0..M_-1
  int tid   = threadIdx.x;              // c channel
  int b     = token / NN_;
  __shared__ float qn[C_];
  __shared__ float red[C_];
  size_t row = (size_t)token * C_;

  int hH = tid >> 5, d = tid & 31;      // wave w == head w (wave32)
  float qv = Pq[row + tid];
  // head-wise max
  red[tid] = qv; __syncthreads();
  for (int s = 16; s > 0; s >>= 1) {
    if (d < s) red[tid] = fmaxf(red[tid], red[tid + s]);
    __syncthreads();
  }
  float mx = red[hH * 32]; __syncthreads();
  float ex = __expf(qv - mx);
  red[tid] = ex; __syncthreads();
  for (int s = 16; s > 0; s >>= 1) {
    if (d < s) red[tid] += red[tid + s];
    __syncthreads();
  }
  float sm = red[hH * 32]; __syncthreads();
  qn[tid] = ex / sm; __syncthreads();

  // attn[c = h*32+e] = sum_d qn[h*32+d] * ctx[b,h,d,e]
  const float* cb = ctx + ((size_t)(b * NH_ + hH) * DH_) * DH_ + d;  // e == d slot
  float acc = 0.f;
#pragma unroll 8
  for (int dd = 0; dd < DH_; ++dd) acc = fmaf(qn[hH * 32 + dd], cb[dd * DH_], acc);

  float xv = inq[row + tid] + acc;
  // LayerNorm over C_
  red[tid] = xv; __syncthreads();
  for (int s = 128; s > 0; s >>= 1) { if (tid < s) red[tid] += red[tid + s]; __syncthreads(); }
  float mean = red[0] * (1.0f / C_); __syncthreads();
  float dv = xv - mean;
  red[tid] = dv * dv; __syncthreads();
  for (int s = 128; s > 0; s >>= 1) { if (tid < s) red[tid] += red[tid + s]; __syncthreads(); }
  float var = red[0] * (1.0f / C_);
  float y = dv * rsqrtf(var + LN_EPS) * g[tid] + bta[tid];
  X[row + tid]   = y;
  X16[row + tid] = (_Float16)y;
}

// ---------------- residual + LN2 -> output ----------------
__global__ __launch_bounds__(256) void ln2_k(
    const float* __restrict__ X, const float* __restrict__ FF,
    const float* __restrict__ g, const float* __restrict__ bta,
    float* __restrict__ out) {
  int token = blockIdx.x;
  int tid   = threadIdx.x;
  __shared__ float red[C_];
  size_t row = (size_t)token * C_;
  float xv = X[row + tid] + FF[row + tid];
  red[tid] = xv; __syncthreads();
  for (int s = 128; s > 0; s >>= 1) { if (tid < s) red[tid] += red[tid + s]; __syncthreads(); }
  float mean = red[0] * (1.0f / C_); __syncthreads();
  float dv = xv - mean;
  red[tid] = dv * dv; __syncthreads();
  for (int s = 128; s > 0; s >>= 1) { if (tid < s) red[tid] += red[tid + s]; __syncthreads(); }
  float var = red[0] * (1.0f / C_);
  out[row + tid] = dv * rsqrtf(var + LN_EPS) * g[tid] + bta[tid];
}

// ---------------- host launch ----------------
extern "C" void kernel_launch(void* const* d_in, const int* in_sizes, int n_in,
                              void* d_out, int out_size, void* d_ws, size_t ws_size,
                              hipStream_t stream) {
  const float* q    = (const float*)d_in[0];
  const float* kv   = (const float*)d_in[1];
  const float* wq_w = (const float*)d_in[2];
  const float* wq_b = (const float*)d_in[3];
  const float* wk_w = (const float*)d_in[4];
  const float* wk_b = (const float*)d_in[5];
  const float* wv_w = (const float*)d_in[6];
  const float* wv_b = (const float*)d_in[7];
  const float* ln1g = (const float*)d_in[8];
  const float* ln1b = (const float*)d_in[9];
  const float* ff1w = (const float*)d_in[10];
  const float* ff1b = (const float*)d_in[11];
  const float* ff2w = (const float*)d_in[12];
  const float* ff2b = (const float*)d_in[13];
  const float* ln2g = (const float*)d_in[14];
  const float* ln2b = (const float*)d_in[15];
  float* out = (float*)d_out;

  // workspace layout (≈330 MB, with reuse)
  const size_t MB = 1ull << 20;
  char* ws = (char*)d_ws;
  _Float16* wqT  = (_Float16*)(ws + 0 * MB);            // 256x256
  _Float16* wkT  = (_Float16*)(ws + 0 * MB + 131072);
  _Float16* wvT  = (_Float16*)(ws + 0 * MB + 262144);
  _Float16* ff1T = (_Float16*)(ws + 0 * MB + 393216);   // 1024x256
  _Float16* ff2T = (_Float16*)(ws + 0 * MB + 917504);   // 256x1024
  _Float16* A16q  = (_Float16*)(ws + 2 * MB);           // M x C f16 (later reused as X16)
  _Float16* A16kv = (_Float16*)(ws + 34 * MB);
  float* Pq = (float*)(ws + 66 * MB);                   // M x C f32 (later reused as FF)
  float* Pk = (float*)(ws + 130 * MB);
  float* Pv = (float*)(ws + 194 * MB);
  _Float16* H16 = (_Float16*)(ws + 130 * MB);           // M x 4C f16, reuses Pk/Pv (134MB)
  float* colmax = (float*)(ws + 264 * MB);
  float* colsum = (float*)(ws + 264 * MB + 4096);
  float* ctxraw = (float*)(ws + 264 * MB + 8192);
  float* ctx    = (float*)(ws + 264 * MB + 131072);
  float* X      = (float*)(ws + 266 * MB);              // M x C f32
  float* FF     = Pq;                                   // M x C f32, reuses Pq
  _Float16* X16 = A16q;                                 // reuses A16q

  // 1) weights -> transposed f16
  transpose_cvt<<<(C_ * C_ + 255) / 256, 256, 0, stream>>>(wq_w, wqT, C_, C_);
  transpose_cvt<<<(C_ * C_ + 255) / 256, 256, 0, stream>>>(wk_w, wkT, C_, C_);
  transpose_cvt<<<(C_ * C_ + 255) / 256, 256, 0, stream>>>(wv_w, wvT, C_, C_);
  transpose_cvt<<<(C_ * FC_ + 255) / 256, 256, 0, stream>>>(ff1w, ff1T, C_, FC_);
  transpose_cvt<<<(FC_ * C_ + 255) / 256, 256, 0, stream>>>(ff2w, ff2T, FC_, C_);
  // activations -> f16
  cvt_f32_f16<<<4096, 256, 0, stream>>>(q,  A16q,  (size_t)M_ * C_);
  cvt_f32_f16<<<4096, 256, 0, stream>>>(kv, A16kv, (size_t)M_ * C_);
  init_stats<<<64, 256, 0, stream>>>(colmax, colsum, ctxraw);

  // 2) QKV projections (async-LDS WMMA f16 -> f32 + bias)
  dim3 gp(M_ / 128, C_ / 64);
  gemm_wmma<0><<<gp, 256, 0, stream>>>(A16q,  wqT, wq_b, Pq, nullptr, C_, C_);
  gemm_wmma<0><<<gp, 256, 0, stream>>>(A16kv, wkT, wk_b, Pk, nullptr, C_, C_);
  gemm_wmma<0><<<gp, 256, 0, stream>>>(A16kv, wvT, wv_b, Pv, nullptr, C_, C_);

  // 3) K column softmax stats + context
  colmax_k<<<B_ * (NN_ / 128), 256, 0, stream>>>(Pk, colmax);
  ctx_accum<<<B_ * NH_ * (NN_ / 512), 256, 0, stream>>>(Pk, Pv, colmax, colsum, ctxraw);
  ctx_final<<<(B_ * NH_ * DH_ * DH_ + 255) / 256, 256, 0, stream>>>(ctxraw, colsum, ctx);

  // 4) q softmax + apply context + residual + LN1
  attn_ln1<<<M_, 256, 0, stream>>>(q, Pq, ctx, ln1g, ln1b, X, X16);

  // 5) FFN GEMM1 (fused exact GELU -> f16)
  dim3 g1(M_ / 128, FC_ / 64);
  gemm_wmma<1><<<g1, 256, 0, stream>>>(X16, ff1T, ff1b, nullptr, H16, FC_, C_);
  // 6) FFN GEMM2 -> f32
  dim3 g2(M_ / 128, C_ / 64);
  gemm_wmma<0><<<g2, 256, 0, stream>>>(H16, ff2T, ff2b, FF, nullptr, C_, FC_);

  // 7) residual + LN2 -> out
  ln2_k<<<M_, 256, 0, stream>>>(X, FF, ln2g, ln2b, out);
}
